// RNNSearch_79224966742174
// MI455X (gfx1250) — matrix operand order repaired and checked by reading, not compiled, not run on hardware
//
#include <hip/hip_runtime.h>
#include <hip/hip_bf16.h>

// =====================================================================
// RNNSearch forward (loss + weighted loss) for gfx1250 / MI455X.
//
// Strategy (compile-only, reasoned from MI455X specs):
//  * All GEMMs run on v_wmma_f32_16x16x32_f16 (f16 in, f32 accumulate).
//  * Vocab GEMM (31 GFLOP, dominant) is fused with online logsumexp so
//    the 242MB logits tensor is never materialized (HBM = 23.3 TB/s but
//    why spend it); affine weights (15.4MB f16) stay L2-resident (192MB).
//  * Recurrent GRU steps are single-workgroup fused kernels: WMMA h@Whh^T
//    into 192KB dynamic LDS (<=320KB/WGP on CDNA5), then gates in-kernel.
//  * Input-side GRU GEMMs are hoisted out of the time loop (M=2048/2016).
//
// Input flattening assumption (JAX pytree = sorted dict keys):
//  0 f_trg[i32 32x64] 1 f_trg_mask[f32 32x64]
//  2 a2o.W[1,256] 3 a2o.b[1] 4 affine.W[30000,256] 5 affine.b[30000]
//  6..9  bigru_b {Whh,Wih,bhh,bih}   10..13 bigru_f {Whh,Wih,bhh,bih}
//  14 c2o.W 15 c2o.b 16 dec_emb[30000,256]
//  17..20 dec_gru1 {Whh,Wih,bhh,bih} 21..24 dec_gru2 {...}
//  25 e2o.W 26 e2o.b 27 enc_emb[30000,256]
//  28..31 enc_gru1 {Whh,Wih[768,512],bhh,bih}
//  32..35 enc_gru2_l0  36..39 enc_gru2_l1
//  40 h2o.W 41 h2o.b 42 h2s.W 43 h2s.b 44 init_affine.W 45 init_affine.b
//  46 s2s.W 47 s2s.b 48 src[i32 32x64] 49 src_mask[f32 32x64]
// =====================================================================

typedef _Float16 v8h  __attribute__((ext_vector_type(8)));
typedef _Float16 v16h __attribute__((ext_vector_type(16)));
typedef float    v8f  __attribute__((ext_vector_type(8)));

#define NEG_INF (-__builtin_inff())

__device__ __forceinline__ float sigmoidf_(float x) { return 1.0f / (1.0f + __expf(-x)); }

// ---- WMMA fragment loaders (layouts per CDNA5 ISA 7.12.2, wave32) ----
// A matrix 16x32 f16: lanes 0-15 row M=lane, K = {0..7,16..23}; lanes 16-31
// row M=lane-16, K = {8..15,24..31}.  Two b128 loads per lane.
__device__ __forceinline__ v16h load_a_frag(const _Float16* __restrict__ base,
                                            int ld, int row0, int k0) {
  int lane = threadIdx.x & 31;
  int g = lane >> 4, r = lane & 15;
  const _Float16* p = base + (size_t)(row0 + r) * ld + k0 + g * 8;
  v8h lo = *(const v8h*)(p);
  v8h hi = *(const v8h*)(p + 16);
  return __builtin_shufflevector(lo, hi, 0,1,2,3,4,5,6,7,8,9,10,11,12,13,14,15);
}

// B matrix 32x16 f16 (we hold W as [N,K] row-major; B[k][n] = W[n][k]):
// lanes 0-15 col N=lane, K = {0..7,16..23}; lanes 16-31 col N=lane-16, K = {8..15,24..31}.
__device__ __forceinline__ v16h load_b_frag(const _Float16* __restrict__ base,
                                            int ld, int col0, int k0) {
  int lane = threadIdx.x & 31;
  int g = lane >> 4, r = lane & 15;
  const _Float16* p = base + (size_t)(col0 + r) * ld + k0 + g * 8;
  v8h lo = *(const v8h*)(p);
  v8h hi = *(const v8h*)(p + 16);
  return __builtin_shufflevector(lo, hi, 0,1,2,3,4,5,6,7,8,9,10,11,12,13,14,15);
}

__device__ __forceinline__ v8f wmma_f32(v16h a, v16h b, v8f c) {
  return __builtin_amdgcn_wmma_f32_16x16x32_f16(false, a, false, b, (short)0, c,
                                                false, false);
}

// ---------------------------------------------------------------------
// Generic WMMA GEMM:  out[M,N] = act( A@W^T (+ A2@W2^T) + bias(+bias2)
//                                     (+ addend) ),  f16 in / f32 out.
// Block = 256 thr = 8 waves in 2x4 covering a 32x64 output tile.
// ---------------------------------------------------------------------
__global__ void gemm_wmma_kernel(const _Float16* __restrict__ A,
                                 const _Float16* __restrict__ W,
                                 const float* __restrict__ bias,
                                 const _Float16* __restrict__ A2,
                                 const _Float16* __restrict__ W2,
                                 const float* __restrict__ bias2,
                                 const float* __restrict__ addend,
                                 float* __restrict__ out,
                                 _Float16* __restrict__ out_h,
                                 int M, int N, int K, int K2, int act_tanh) {
  int w  = threadIdx.x >> 5;
  int wm = w >> 2, wn = w & 3;
  int tM = blockIdx.y * 32 + wm * 16;
  int tN = blockIdx.x * 64 + wn * 16;
  if (tM >= M || tN >= N) return;  // wave-uniform
  v8f acc = {};
  for (int k = 0; k < K; k += 32)
    acc = wmma_f32(load_a_frag(A, K, tM, k), load_b_frag(W, K, tN, k), acc);
  if (A2)
    for (int k = 0; k < K2; k += 32)
      acc = wmma_f32(load_a_frag(A2, K2, tM, k), load_b_frag(W2, K2, tN, k), acc);
  int lane = threadIdx.x & 31;
  int g = lane >> 4, r = lane & 15;
  int col = tN + r;
  float bsum = (bias ? bias[col] : 0.0f) + (bias2 ? bias2[col] : 0.0f);
#pragma unroll
  for (int i = 0; i < 8; i++) {
    int row = tM + i + 8 * g;
    float v = acc[i] + bsum;
    if (addend) v += addend[(size_t)row * N + col];
    if (act_tanh) v = tanhf(v);
    out[(size_t)row * N + col] = v;
    if (out_h) out_h[(size_t)row * N + col] = (_Float16)v;
  }
}

// ---------------------------------------------------------------------
// Fused GRU step. One workgroup (8 waves). Computes gh = h@Whh^T (and gi
// = x@Wih^T when not precomputed) via WMMA into LDS, then gates.
// Dynamic LDS: 2 * 32*768 floats = 192KB (CDNA5 WGP has 320KB).
// ---------------------------------------------------------------------
__global__ void gru_step_kernel(const float* __restrict__ gi_pre,   // [32,768] incl bih, or null
                                const _Float16* __restrict__ x_h,   // [32,Kx]  if gi_pre==null
                                const _Float16* __restrict__ Wih_h, // [768,Kx]
                                const float* __restrict__ bih,      // [768]
                                const _Float16* __restrict__ Whh_h, // [768,256]
                                const float* __restrict__ bhh,      // [768]
                                float* __restrict__ hid,            // [32,256] in/out
                                _Float16* __restrict__ hid_h,       // [32,256] in/out
                                float* __restrict__ ys,             // optional [32,*]
                                _Float16* __restrict__ ys_h,        // optional
                                int ys_ld, int Kx) {
  extern __shared__ float lds[];
  float* gh = lds;              // [32][768]
  float* gi = lds + 32 * 768;   // [32][768]
  int w = threadIdx.x >> 5;
  int lane = threadIdx.x & 31;
  int g = lane >> 4, r = lane & 15;

  for (int tix = w; tix < 96; tix += 8) {   // 2 M-tiles x 48 N-tiles
    int tM = (tix / 48) * 16;
    int tN = (tix % 48) * 16;
    v8f acc = {};
    for (int k = 0; k < 256; k += 32)
      acc = wmma_f32(load_a_frag(hid_h, 256, tM, k),
                     load_b_frag(Whh_h, 256, tN, k), acc);
#pragma unroll
    for (int i = 0; i < 8; i++)
      gh[(tM + i + 8 * g) * 768 + tN + r] = acc[i];
    if (!gi_pre) {
      v8f acc2 = {};
      for (int k = 0; k < Kx; k += 32)
        acc2 = wmma_f32(load_a_frag(x_h, Kx, tM, k),
                        load_b_frag(Wih_h, Kx, tN, k), acc2);
#pragma unroll
      for (int i = 0; i < 8; i++)
        gi[(tM + i + 8 * g) * 768 + tN + r] = acc2[i];
    }
  }
  __syncthreads();

  for (int idx = threadIdx.x; idx < 32 * 256; idx += 256) {
    int b = idx >> 8, j = idx & 255;
    float gir, giz, gin;
    if (gi_pre) {
      gir = gi_pre[b * 768 + j];
      giz = gi_pre[b * 768 + 256 + j];
      gin = gi_pre[b * 768 + 512 + j];
    } else {
      gir = gi[b * 768 + j] + bih[j];
      giz = gi[b * 768 + 256 + j] + bih[256 + j];
      gin = gi[b * 768 + 512 + j] + bih[512 + j];
    }
    float ghr = gh[b * 768 + j] + bhh[j];
    float ghz = gh[b * 768 + 256 + j] + bhh[256 + j];
    float ghn = gh[b * 768 + 512 + j] + bhh[512 + j];
    float rr = sigmoidf_(gir + ghr);
    float zz = sigmoidf_(giz + ghz);
    float nn = tanhf(gin + rr * ghn);
    float hprev = hid[idx];
    float hnew = (1.0f - zz) * nn + zz * hprev;
    hid[idx] = hnew;
    hid_h[idx] = (_Float16)hnew;
    if (ys)   ys[b * ys_ld + j] = hnew;
    if (ys_h) ys_h[b * ys_ld + j] = (_Float16)hnew;
  }
}

// ---------------------------------------------------------------------
// Attention for one decoder step. Block per batch element b.
// ---------------------------------------------------------------------
__global__ void attn_kernel(const float* __restrict__ ctx_proj,  // [S,B,256]
                            const float* __restrict__ hs,        // [32,256]
                            const float* __restrict__ a2o_w,     // [256]
                            const float* __restrict__ a2o_b,     // [1]
                            const float* __restrict__ src_mask,  // [B,S]
                            const float* __restrict__ ctx,       // [S,B,256]
                            float* __restrict__ attn,            // [32,256]
                            _Float16* __restrict__ attn_h) {
  __shared__ float lg[64];
  __shared__ float wsm[64];
  int b = blockIdx.x;
  int w = threadIdx.x >> 5, lane = threadIdx.x & 31;
  for (int s = w; s < 64; s += 8) {
    float p = 0.0f;
#pragma unroll
    for (int j = 0; j < 8; j++) {
      int d = lane + 32 * j;
      float a = tanhf(ctx_proj[(size_t)s * 8192 + b * 256 + d] + hs[b * 256 + d]);
      p += a * a2o_w[d];
    }
    for (int m = 16; m >= 1; m >>= 1) p += __shfl_xor(p, m, 32);
    if (lane == 0) lg[s] = p + a2o_b[0];
  }
  __syncthreads();
  if (w == 0) {
    float v0 = (src_mask[b * 64 + lane]      > 0.0f) ? lg[lane]      : NEG_INF;
    float v1 = (src_mask[b * 64 + lane + 32] > 0.0f) ? lg[lane + 32] : NEG_INF;
    float mx = fmaxf(v0, v1);
    for (int m = 16; m >= 1; m >>= 1) mx = fmaxf(mx, __shfl_xor(mx, m, 32));
    float e0 = __expf(v0 - mx), e1 = __expf(v1 - mx);
    float sm = e0 + e1;
    for (int m = 16; m >= 1; m >>= 1) sm += __shfl_xor(sm, m, 32);
    wsm[lane] = e0 / sm;
    wsm[lane + 32] = e1 / sm;
  }
  __syncthreads();
  int d = threadIdx.x;
  float acc = 0.0f;
  for (int s = 0; s < 64; s++)
    acc += wsm[s] * ctx[(size_t)s * 8192 + b * 256 + d];
  attn[b * 256 + d] = acc;
  attn_h[b * 256 + d] = (_Float16)acc;
}

// ---------------------------------------------------------------------
// Vocab GEMM fused with online logsumexp + target-logit pick.
// Grid: 126 blocks (16 rows each), 128 threads (4 waves). Each wave
// streams N-chunks of 16 vocab columns; WMMA K=256 in 8 steps.
// ---------------------------------------------------------------------
__global__ void vocab_nll_kernel(const _Float16* __restrict__ outs_h, // [2016,256]
                                 const _Float16* __restrict__ Wv,     // [30000,256]
                                 const float* __restrict__ bv,        // [30000]
                                 const int* __restrict__ f_trg,       // [32,64]
                                 float* __restrict__ nll) {           // [2016]
  __shared__ float rm[16][65], rs[16][65], rt[16][65];
  int m0 = blockIdx.x * 16;
  int w = threadIdx.x >> 5, lane = threadIdx.x & 31;
  int g = lane >> 4, r = lane & 15;

  v16h af[8];
#pragma unroll
  for (int kk = 0; kk < 8; kk++) af[kk] = load_a_frag(outs_h, 256, m0, kk * 32);

  int tgt[8];
#pragma unroll
  for (int i = 0; i < 8; i++) {
    int row = m0 + i + 8 * g;      // [0,2016): row = t*32 + b
    int t = row >> 5, b = row & 31;
    tgt[i] = f_trg[b * 64 + t + 1];
  }
  float mrun[8], srun[8], tl[8];
#pragma unroll
  for (int i = 0; i < 8; i++) { mrun[i] = NEG_INF; srun[i] = 0.0f; tl[i] = NEG_INF; }

  for (int c = w; c < 1875; c += 4) {    // 30000/16 column chunks
    int n0 = c * 16;
    v8f acc = {};
#pragma unroll
    for (int kk = 0; kk < 8; kk++)
      acc = wmma_f32(af[kk], load_b_frag(Wv, 256, n0, kk * 32), acc);
    int col = n0 + r;
    float bb = bv[col];
#pragma unroll
    for (int i = 0; i < 8; i++) {
      float L = acc[i] + bb;
      float nm = fmaxf(mrun[i], L);
      srun[i] = srun[i] * __expf(mrun[i] - nm) + __expf(L - nm);
      mrun[i] = nm;
      if (col == tgt[i]) tl[i] = L;
    }
  }
#pragma unroll
  for (int i = 0; i < 8; i++) {
    rm[i + 8 * g][w * 16 + r] = mrun[i];
    rs[i + 8 * g][w * 16 + r] = srun[i];
    rt[i + 8 * g][w * 16 + r] = tl[i];
  }
  __syncthreads();
  if (threadIdx.x < 16) {
    int row = threadIdx.x;
    float M = NEG_INF, S = 0.0f, T = NEG_INF;
    for (int k = 0; k < 64; k++) {
      float m2 = rm[row][k], s2 = rs[row][k];
      float nm = fmaxf(M, m2);
      S = S * __expf(M - nm) + s2 * __expf(m2 - nm);
      M = nm;
      T = fmaxf(T, rt[row][k]);
    }
    nll[m0 + row] = (M + __logf(S)) - T;   // lse - target logit
  }
}

// ---------------------------------------------------------------------
__global__ void loss_kernel(const float* __restrict__ nll,
                            const float* __restrict__ f_trg_mask,
                            float* __restrict__ out) {
  __shared__ float s1[256], s2[256];
  float a = 0.0f, mm = 0.0f;
  for (int idx = threadIdx.x; idx < 2016; idx += 256) {
    int t = idx >> 5, b = idx & 31;
    float msk = f_trg_mask[b * 64 + t + 1];
    a += nll[idx] * msk;
    mm += msk;
  }
  s1[threadIdx.x] = a; s2[threadIdx.x] = mm;
  __syncthreads();
  for (int st = 128; st > 0; st >>= 1) {
    if (threadIdx.x < st) {
      s1[threadIdx.x] += s1[threadIdx.x + st];
      s2[threadIdx.x] += s2[threadIdx.x + st];
    }
    __syncthreads();
  }
  if (threadIdx.x == 0) {
    out[0] = s1[0] / 32.0f;   // loss  = mean_b(sum_t nll*mask)
    out[1] = s1[0] / s2[0];   // w_loss
  }
}

// ---------------------------------------------------------------------
__global__ void gather_emb_kernel(const int* __restrict__ tok,
                                  const float* __restrict__ emb,
                                  float* __restrict__ out,
                                  _Float16* __restrict__ out_h,
                                  int B, int D, int total) {
  int i = blockIdx.x * blockDim.x + threadIdx.x;
  if (i >= total) return;
  int d = i % D;
  int b = (i / D) % B;
  int t = i / (D * B);
  int v = tok[b * 64 + t];
  float val = emb[(size_t)v * D + d];
  out[i] = val;
  out_h[i] = (_Float16)val;
}

__global__ void f32_to_f16_kernel(const float* __restrict__ in,
                                  _Float16* __restrict__ out, int n) {
  int i = blockIdx.x * blockDim.x + threadIdx.x;
  if (i < n) out[i] = (_Float16)in[i];
}

__global__ void avgctx_kernel(const float* __restrict__ ctx,       // [S,B,256]
                              const float* __restrict__ src_mask,  // [B,S]
                              float* __restrict__ avg,
                              _Float16* __restrict__ avg_h) {
  int b = blockIdx.x, d = threadIdx.x;
  float s = 0.0f;
  for (int t = 0; t < 64; t++) s += ctx[(size_t)t * 8192 + b * 256 + d];
  float den = 0.0f;
  for (int t = 0; t < 64; t++) den += src_mask[b * 64 + t];
  float v = s / den;
  avg[b * 256 + d] = v;
  avg_h[b * 256 + d] = (_Float16)v;
}

// =====================================================================
extern "C" void kernel_launch(void* const* d_in, const int* in_sizes, int n_in,
                              void* d_out, int out_size, void* d_ws, size_t ws_size,
                              hipStream_t stream) {
  (void)in_sizes; (void)n_in; (void)out_size; (void)ws_size;
  const int S = 64, B = 32, Td = 63;

  // ---- inputs (see flattening assumption at top) ----
  const int*   f_trg      = (const int*)  d_in[0];
  const float* f_trg_mask = (const float*)d_in[1];
  const float* a2o_W = (const float*)d_in[2];
  const float* a2o_b = (const float*)d_in[3];
  const float* aff_W = (const float*)d_in[4];
  const float* aff_b = (const float*)d_in[5];
  struct GruP { const float *Whh, *Wih, *bhh, *bih; };
  auto gru_at = [&](int i) {
    return GruP{(const float*)d_in[i], (const float*)d_in[i + 1],
                (const float*)d_in[i + 2], (const float*)d_in[i + 3]};
  };
  GruP gb  = gru_at(6),  gf  = gru_at(10);
  const float* c2o_W = (const float*)d_in[14];
  const float* c2o_b = (const float*)d_in[15];
  const float* dec_emb = (const float*)d_in[16];
  GruP gd1 = gru_at(17), gd2 = gru_at(21);
  const float* e2o_W = (const float*)d_in[25];
  const float* e2o_b = (const float*)d_in[26];
  const float* enc_emb = (const float*)d_in[27];
  GruP ge1 = gru_at(28), gl0 = gru_at(32), gl1 = gru_at(36);
  const float* h2o_W  = (const float*)d_in[40];
  const float* h2o_b  = (const float*)d_in[41];
  const float* h2s_W  = (const float*)d_in[42];
  const float* h2s_b  = (const float*)d_in[43];
  const float* init_W = (const float*)d_in[44];
  const float* init_b = (const float*)d_in[45];
  const float* s2s_W  = (const float*)d_in[46];
  const float* s2s_b  = (const float*)d_in[47];
  const int*   src    = (const int*)  d_in[48];
  const float* src_mask = (const float*)d_in[49];

  // ---- workspace bump allocator ----
  char* cur = (char*)d_ws;
  auto alloc = [&](size_t bytes) -> void* {
    void* p = (void*)cur;
    cur += (bytes + 255) & ~(size_t)255;
    return p;
  };
  typedef _Float16 h16;
  // f16 weight copies
  h16* wf_ih = (h16*)alloc(768 * 256 * 2); h16* wf_hh = (h16*)alloc(768 * 256 * 2);
  h16* wb_ih = (h16*)alloc(768 * 256 * 2); h16* wb_hh = (h16*)alloc(768 * 256 * 2);
  h16* e1_ih = (h16*)alloc(768 * 512 * 2); h16* e1_hh = (h16*)alloc(768 * 256 * 2);
  h16* l0_ih = (h16*)alloc(768 * 256 * 2); h16* l0_hh = (h16*)alloc(768 * 256 * 2);
  h16* l1_ih = (h16*)alloc(768 * 256 * 2); h16* l1_hh = (h16*)alloc(768 * 256 * 2);
  h16* d1_ih = (h16*)alloc(768 * 256 * 2); h16* d1_hh = (h16*)alloc(768 * 256 * 2);
  h16* d2_ih = (h16*)alloc(768 * 256 * 2); h16* d2_hh = (h16*)alloc(768 * 256 * 2);
  h16* h2s_h = (h16*)alloc(65536 * 2); h16* s2s_h = (h16*)alloc(65536 * 2);
  h16* e2o_h = (h16*)alloc(65536 * 2); h16* h2o_h = (h16*)alloc(65536 * 2);
  h16* c2o_h = (h16*)alloc(65536 * 2); h16* ini_h = (h16*)alloc(65536 * 2);
  h16* aff_h = (h16*)alloc((size_t)30000 * 256 * 2);
  // activations
  float* x_enc   = (float*)alloc(2048 * 256 * 4); h16* x_enc_h = (h16*)alloc(2048 * 256 * 2);
  float* gi_big  = (float*)alloc(2048 * 768 * 4);
  float* hcat    = (float*)alloc(2048 * 512 * 4); h16* hcat_h  = (h16*)alloc(2048 * 512 * 2);
  float* bufA    = (float*)alloc(2048 * 256 * 4); h16* bufA_h  = (h16*)alloc(2048 * 256 * 2);
  float* bufB    = (float*)alloc(2048 * 256 * 4); h16* bufB_h  = (h16*)alloc(2048 * 256 * 2);
  float* ctx     = (float*)alloc(2048 * 256 * 4); h16* ctx_h   = (h16*)alloc(2048 * 256 * 2);
  float* ctxp    = (float*)alloc(2048 * 256 * 4);
  float* avg     = (float*)alloc(32 * 256 * 4);   h16* avg_h   = (h16*)alloc(32 * 256 * 2);
  float* hid     = (float*)alloc(32 * 256 * 4);   h16* hid_h   = (h16*)alloc(32 * 256 * 2);
  float* e_dec   = (float*)alloc(2016 * 256 * 4); h16* e_h     = (h16*)alloc(2016 * 256 * 2);
  float* Eo      = (float*)alloc(2016 * 256 * 4);
  float* hs      = (float*)alloc(32 * 256 * 4);
  float* attnv   = (float*)alloc(32 * 256 * 4);   h16* attnv_h = (h16*)alloc(32 * 256 * 2);
  float* outs    = (float*)alloc(2016 * 256 * 4); h16* outs_h  = (h16*)alloc(2016 * 256 * 2);
  float* nll     = (float*)alloc(2016 * 4);

  auto conv = [&](const float* s, h16* dst, int n) {
    f32_to_f16_kernel<<<(n + 255) / 256, 256, 0, stream>>>(s, dst, n);
  };
  auto gemm = [&](const h16* A, const h16* W, const float* bias,
                  const h16* A2, const h16* W2, const float* bias2,
                  const float* addend, float* out, h16* out_h16,
                  int M, int N, int K, int K2, int act) {
    dim3 grid(N / 64, M / 32);
    gemm_wmma_kernel<<<grid, 256, 0, stream>>>(A, W, bias, A2, W2, bias2, addend,
                                               out, out_h16, M, N, K, K2, act);
  };
  const size_t GRU_LDS = 2 * 32 * 768 * sizeof(float);  // 192KB (<=320KB/WGP)
  auto gru_step = [&](const float* gi_pre, const h16* x_h, const h16* Wih,
                      const float* bih, const h16* Whh, const float* bhh,
                      float* ys, h16* ys_h, int ys_ld) {
    gru_step_kernel<<<1, 256, GRU_LDS, stream>>>(gi_pre, x_h, Wih, bih, Whh, bhh,
                                                 hid, hid_h, ys, ys_h, ys_ld, 256);
  };
  auto reset_hid = [&]() {
    hipMemsetAsync(hid, 0, 32 * 256 * 4, stream);
    hipMemsetAsync(hid_h, 0, 32 * 256 * 2, stream);
  };

  // ---- weight conversions (once per call; deterministic) ----
  conv(gf.Wih, wf_ih, 768 * 256); conv(gf.Whh, wf_hh, 768 * 256);
  conv(gb.Wih, wb_ih, 768 * 256); conv(gb.Whh, wb_hh, 768 * 256);
  conv(ge1.Wih, e1_ih, 768 * 512); conv(ge1.Whh, e1_hh, 768 * 256);
  conv(gl0.Wih, l0_ih, 768 * 256); conv(gl0.Whh, l0_hh, 768 * 256);
  conv(gl1.Wih, l1_ih, 768 * 256); conv(gl1.Whh, l1_hh, 768 * 256);
  conv(gd1.Wih, d1_ih, 768 * 256); conv(gd1.Whh, d1_hh, 768 * 256);
  conv(gd2.Wih, d2_ih, 768 * 256); conv(gd2.Whh, d2_hh, 768 * 256);
  conv(h2s_W, h2s_h, 65536); conv(s2s_W, s2s_h, 65536);
  conv(e2o_W, e2o_h, 65536); conv(h2o_W, h2o_h, 65536);
  conv(c2o_W, c2o_h, 65536); conv(init_W, ini_h, 65536);
  conv(aff_W, aff_h, 30000 * 256);

  // ---- encoder ----
  gather_emb_kernel<<<(2048 * 256 + 255) / 256, 256, 0, stream>>>(
      src, enc_emb, x_enc, x_enc_h, B, 256, 2048 * 256);

  // forward GRU: precompute gi for whole sequence, then recur
  gemm(x_enc_h, wf_ih, gf.bih, nullptr, nullptr, nullptr, nullptr,
       gi_big, nullptr, 2048, 768, 256, 0, 0);
  reset_hid();
  for (int t = 0; t < S; t++)
    gru_step(gi_big + (size_t)t * 32 * 768, nullptr, nullptr, nullptr, wf_hh, gf.bhh,
             hcat + (size_t)t * 32 * 512, hcat_h + (size_t)t * 32 * 512, 512);
  // backward GRU (processes reversed sequence; writes to original position)
  gemm(x_enc_h, wb_ih, gb.bih, nullptr, nullptr, nullptr, nullptr,
       gi_big, nullptr, 2048, 768, 256, 0, 0);
  reset_hid();
  for (int t = 0; t < S; t++) {
    int s = S - 1 - t;
    gru_step(gi_big + (size_t)s * 32 * 768, nullptr, nullptr, nullptr, wb_hh, gb.bhh,
             hcat + (size_t)s * 32 * 512 + 256, hcat_h + (size_t)s * 32 * 512 + 256, 512);
  }
  // enc_gru1 (input 2H=512)
  gemm(hcat_h, e1_ih, ge1.bih, nullptr, nullptr, nullptr, nullptr,
       gi_big, nullptr, 2048, 768, 512, 0, 0);
  reset_hid();
  for (int t = 0; t < S; t++)
    gru_step(gi_big + (size_t)t * 32 * 768, nullptr, nullptr, nullptr, e1_hh, ge1.bhh,
             bufA + (size_t)t * 32 * 256, bufA_h + (size_t)t * 32 * 256, 256);
  // enc_gru2_l0
  gemm(bufA_h, l0_ih, gl0.bih, nullptr, nullptr, nullptr, nullptr,
       gi_big, nullptr, 2048, 768, 256, 0, 0);
  reset_hid();
  for (int t = 0; t < S; t++)
    gru_step(gi_big + (size_t)t * 32 * 768, nullptr, nullptr, nullptr, l0_hh, gl0.bhh,
             bufB + (size_t)t * 32 * 256, bufB_h + (size_t)t * 32 * 256, 256);
  // enc_gru2_l1 -> enc_context
  gemm(bufB_h, l1_ih, gl1.bih, nullptr, nullptr, nullptr, nullptr,
       gi_big, nullptr, 2048, 768, 256, 0, 0);
  reset_hid();
  for (int t = 0; t < S; t++)
    gru_step(gi_big + (size_t)t * 32 * 768, nullptr, nullptr, nullptr, l1_hh, gl1.bhh,
             ctx + (size_t)t * 32 * 256, ctx_h + (size_t)t * 32 * 256, 256);

  // decoder init hidden = tanh(avg @ init_W^T + b)
  avgctx_kernel<<<32, 256, 0, stream>>>(ctx, src_mask, avg, avg_h);
  gemm(avg_h, ini_h, init_b, nullptr, nullptr, nullptr, nullptr,
       hid, hid_h, 32, 256, 256, 0, 1);
  // ctx_proj = enc_context @ s2s^T + b
  gemm(ctx_h, s2s_h, s2s_b, nullptr, nullptr, nullptr, nullptr,
       ctxp, nullptr, 2048, 256, 256, 0, 0);

  // decoder input embeddings (f_trg[:, :-1]) + hoisted GEMMs
  gather_emb_kernel<<<(2016 * 256 + 255) / 256, 256, 0, stream>>>(
      f_trg, dec_emb, e_dec, e_h, B, 256, 2016 * 256);
  gemm(e_h, e2o_h, e2o_b, nullptr, nullptr, nullptr, nullptr,
       Eo, nullptr, 2016, 256, 256, 0, 0);
  gemm(e_h, d1_ih, gd1.bih, nullptr, nullptr, nullptr, nullptr,
       gi_big, nullptr, 2016, 768, 256, 0, 0);   // reuse gi_big (2016<=2048 rows)

  // ---- decoder time loop ----
  for (int t = 0; t < Td; t++) {
    // h1 = GRU1(e_t, hid)
    gru_step(gi_big + (size_t)t * 32 * 768, nullptr, nullptr, nullptr, d1_hh, gd1.bhh,
             nullptr, nullptr, 256);
    // hs = h1 @ h2s^T + b
    gemm(hid_h, h2s_h, h2s_b, nullptr, nullptr, nullptr, nullptr,
         hs, nullptr, 32, 256, 256, 0, 0);
    // attention
    attn_kernel<<<32, 256, 0, stream>>>(ctxp, hs, a2o_W, a2o_b, src_mask, ctx,
                                        attnv, attnv_h);
    // h2 = GRU2(attn, h1)  (gi computed in-kernel via WMMA)
    gru_step(nullptr, attnv_h, d2_ih, gd2.bih, d2_hh, gd2.bhh, nullptr, nullptr, 256);
    // out_t = tanh(Eo_t + h2@h2o^T + b + attn@c2o^T + b)
    gemm(hid_h, h2o_h, h2o_b, attnv_h, c2o_h, c2o_b, Eo + (size_t)t * 32 * 256,
         outs + (size_t)t * 32 * 256, outs_h + (size_t)t * 32 * 256,
         32, 256, 256, 256, 1);
  }

  // ---- fused vocab GEMM + logsumexp + NLL, then loss reduction ----
  vocab_nll_kernel<<<126, 128, 0, stream>>>(outs_h, aff_h, aff_b, f_trg, nll);
  loss_kernel<<<1, 256, 0, stream>>>(nll, f_trg_mask, (float*)d_out);
}
